// LSTM_56916906607121
// MI455X (gfx1250) — matrix operand order repaired and checked by reading, not compiled, not run on hardware
//
#include <hip/hip_runtime.h>

typedef __attribute__((ext_vector_type(16))) _Float16 v16h;
typedef __attribute__((ext_vector_type(8)))  _Float16 v8h;
typedef __attribute__((ext_vector_type(8)))  float    v8f;
typedef __attribute__((ext_vector_type(4)))  int      v4i;

// global-address-space views for guaranteed global_load_b128 codegen
typedef __attribute__((address_space(1))) const v16h     gv16h;
typedef __attribute__((address_space(1))) const _Float16 gf16;
// async global->LDS builtin operand types (from hipcc diagnostic):
// (int4 addrspace(1)*, int4 addrspace(3)*, imm, imm)
typedef __attribute__((address_space(1))) v4i gv4i;
typedef __attribute__((address_space(3))) v4i lv4i;

#define HDIM 256
#define TSTEPS 256
#define BM 16                       // batch rows per block (WMMA M)
#define HSTR 264                    // h LDS row stride (halves): 132 DW = 4 mod 64 -> conflict-free
#define XSTR 256                    // x LDS row stride (floats)
#define FRAG_HALVES 512             // one 32x16 f16 B fragment
#define KT_STRIDE (16 * FRAG_HALVES)          // 16 nt fragments per kt
#define GATE_STRIDE (8 * 16 * FRAG_HALVES)    // 8 kt x 16 nt fragments per gate

// sched_group_barrier masks (LLVM AMDGPU)
#define SG_MFMA   0x008             // MFMA/WMMA
#define SG_VMEM_R 0x020             // VMEM read
#define SG_DS_R   0x100             // DS read

__device__ __forceinline__ float fsigmoid(float x) {
    return 1.0f / (1.0f + __expf(-x));
}
__device__ __forceinline__ float ftanh(float x) {
    // tanh(x) = 1 - 2/(e^{2x}+1)
    return 1.0f - 2.0f / (__expf(2.0f * x) + 1.0f);
}

// ---------------------------------------------------------------------------
// Prep: convert 4 HxH f32 recurrent weight matrices to f16 B-fragments in the
// contiguous-K per-lane order that matches this kernel's A loads:
//   fragment(q, kt, nt): 32 lanes x 16 halves contiguous;
//   lane L -> column n = nt*16 + (L&15), K = kt*32 + (L>>4)*16 + j (j=0..15)
// ---------------------------------------------------------------------------
__global__ void lstm_prep_weights(const float* __restrict__ Wg,
                                  const float* __restrict__ Wi,
                                  const float* __restrict__ Wf,
                                  const float* __restrict__ Wo,
                                  _Float16* __restrict__ wswz) {
    int flat = blockIdx.x * blockDim.x + threadIdx.x;   // 0 .. 4*H*H-1
    int j    = flat & 15;
    int lane = (flat >> 4) & 31;
    int frag = flat >> 9;
    int nt   = frag & 15;
    int kt   = (frag >> 4) & 7;
    int q    = frag >> 7;
    int k = kt * 32 + (lane >> 4) * 16 + j;
    int n = nt * 16 + (lane & 15);
    const float* W = (q == 0) ? Wg : (q == 1) ? Wi : (q == 2) ? Wf : Wo;
    wswz[flat] = (_Float16)W[k * HDIM + n];
}

// ---------------------------------------------------------------------------
// Persistent LSTM scan. One block = 16 batch rows, 16 waves (512 threads).
// Wave w owns N-tile nt=w (16 columns), all 4 gates, K=256 in 8 chunks.
// h (f16) double-buffered in LDS; c lives in 8 VGPRs per lane; x staged in
// LDS via async global->LDS. The kt pipeline is enforced with
// sched_group_barrier: loads for kt+1 are in flight while WMMAs for kt run.
// Per step per wave: 8 kt x 4 gates = 32 v_wmma_f32_16x16x32_f16.
// ---------------------------------------------------------------------------
__global__ void __launch_bounds__(512) lstm_persistent(
    const float* __restrict__ x,
    const float* __restrict__ Wgx, const float* __restrict__ bg,
    const float* __restrict__ Wix, const float* __restrict__ bi,
    const float* __restrict__ Wfx, const float* __restrict__ bf,
    const float* __restrict__ Wox, const float* __restrict__ bo,
    const float* __restrict__ Wph, const float* __restrict__ bp,
    const _Float16* __restrict__ wswz,
    float* __restrict__ out)
{
    extern __shared__ char smem[];
    _Float16* h0   = (_Float16*)smem;                                   // 8448 B
    _Float16* h1   = (_Float16*)(smem + (size_t)BM * HSTR * 2);         // 8448 B
    float*    xlds = (float*)(smem + (size_t)2 * BM * HSTR * 2);        // 16384 B

    const int tid     = threadIdx.x;
    const int lane    = tid & 31;
    const int nt      = tid >> 5;        // wave id == N-tile id (0..15)
    const int rowbase = blockIdx.x * BM;

    const int lhalf = lane >> 4;         // 0: lanes 0-15, 1: lanes 16-31
    const int lmod  = lane & 15;
    const int rhalf = lhalf * 8;         // D-rows rhalf..rhalf+7 for this lane
    const int aoff0 = lhalf * 16;        // A-fragment K sub-offset
    const int arow  = lmod;              // A-fragment row for this lane

    // --- init: zero h, stage this block's x slice (16 x 256 f32) in LDS ---
    for (int i2 = tid; i2 < BM * HSTR; i2 += 512) h0[i2] = (_Float16)0.0f;

#if defined(__gfx1250__) && __has_builtin(__builtin_amdgcn_global_load_async_to_lds_b128)
    // CDNA5 async global->LDS path (ASYNCcnt-tracked), 16B per lane
    for (int i2 = tid; i2 < (BM * TSTEPS) / 4; i2 += 512) {
        int f4row = i2 >> 6;             // 64 float4 per row
        int f4col = i2 & 63;
        gv4i* gsrc = (gv4i*)(x + (size_t)(rowbase + f4row) * TSTEPS + f4col * 4);
        lv4i* ldst = (lv4i*)(xlds + f4row * XSTR + f4col * 4);
        __builtin_amdgcn_global_load_async_to_lds_b128(gsrc, ldst, 0, 0);
    }
#if __has_builtin(__builtin_amdgcn_s_wait_asynccnt)
    __builtin_amdgcn_s_wait_asynccnt(0);
#else
    asm volatile("s_wait_asynccnt 0x0" ::: "memory");
#endif
#else
    for (int i2 = tid; i2 < BM * TSTEPS; i2 += 512) {
        int row = i2 >> 8;
        int tt  = i2 & 255;
        xlds[row * XSTR + tt] = x[(size_t)(rowbase + row) * TSTEPS + tt];
    }
#endif
    __syncthreads();

    _Float16* hc = h0;
    _Float16* hn = h1;

    // per-lane loop-invariant column scalars (one fixed output column per lane)
    const int ncol = nt * 16 + lmod;
    const float wgx = Wgx[ncol], wix = Wix[ncol];
    const float wfx = Wfx[ncol], wox = Wox[ncol];
    const float bgn = bg[ncol], bin = bi[ncol];
    const float bfn = bf[ncol], bon = bo[ncol];

    // per-lane cell state: rows rhalf..rhalf+7 of column ncol
    float creg[8];
    #pragma unroll
    for (int r = 0; r < 8; ++r) creg[r] = 0.0f;

    // opaque 32-bit offset (always 0): defeats LICM on the weight loads
    // (hoisting 32 fragments would spill to scratch) while keeping the base
    // pointer's address space inferable.
    unsigned int woff = 0;

    for (int t = 0; t < TSTEPS; ++t) {
        asm volatile("" : "+s"(woff));
        const gf16* wbase = (const gf16*)(wswz + woff)
                          + (size_t)nt * FRAG_HALVES + lane * 16;
        const _Float16* abase = hc + arow * HSTR + aoff0;

        v8f ag = {}, ai = {}, af = {}, ao = {};

        // ---- prologue: fragments for kt = 0 ----
        v16h B0g = *(gv16h*)(wbase + 0 * GATE_STRIDE);
        v16h B0i = *(gv16h*)(wbase + 1 * GATE_STRIDE);
        v16h B0f = *(gv16h*)(wbase + 2 * GATE_STRIDE);
        v16h B0o = *(gv16h*)(wbase + 3 * GATE_STRIDE);
        v8h  a0  = *(const v8h*)(abase);
        v8h  a1  = *(const v8h*)(abase + 8);
        v16h A0  = __builtin_shufflevector(a0, a1,
                    0,1,2,3,4,5,6,7,8,9,10,11,12,13,14,15);

        #pragma unroll
        for (int kt = 0; kt < 8; ++kt) {
            // issue next iteration's fragment loads before consuming this one
            v16h B1g = {}, B1i = {}, B1f = {}, B1o = {}, A1 = {};
            if (kt < 7) {
                const gf16* wk = wbase + (size_t)(kt + 1) * KT_STRIDE;
                B1g = *(gv16h*)(wk + 0 * GATE_STRIDE);
                B1i = *(gv16h*)(wk + 1 * GATE_STRIDE);
                B1f = *(gv16h*)(wk + 2 * GATE_STRIDE);
                B1o = *(gv16h*)(wk + 3 * GATE_STRIDE);
                const _Float16* ap = abase + (kt + 1) * 32;
                v8h n0 = *(const v8h*)(ap);
                v8h n1 = *(const v8h*)(ap + 8);
                A1 = __builtin_shufflevector(n0, n1,
                      0,1,2,3,4,5,6,7,8,9,10,11,12,13,14,15);
            }

            ag = __builtin_amdgcn_wmma_f32_16x16x32_f16(false, A0, false, B0g, (short)0, ag, false, false);
            ai = __builtin_amdgcn_wmma_f32_16x16x32_f16(false, A0, false, B0i, (short)0, ai, false, false);
            af = __builtin_amdgcn_wmma_f32_16x16x32_f16(false, A0, false, B0f, (short)0, af, false, false);
            ao = __builtin_amdgcn_wmma_f32_16x16x32_f16(false, A0, false, B0o, (short)0, ao, false, false);

            B0g = B1g; B0i = B1i; B0f = B1f; B0o = B1o; A0 = A1;
        }

#if __has_builtin(__builtin_amdgcn_sched_group_barrier)
        // Pin the software pipeline the post-RA scheduler otherwise undoes:
        //   {loads kt0} {loads kt1} {wmma kt0} {loads kt2} {wmma kt1} ...
        // Each load group: 8 global_load_b128 (B frags) + 2 ds_load_b128 (A).
        __builtin_amdgcn_sched_group_barrier(SG_VMEM_R, 8, 0);
        __builtin_amdgcn_sched_group_barrier(SG_DS_R,   2, 0);
        #pragma unroll
        for (int kt = 0; kt < 7; ++kt) {
            __builtin_amdgcn_sched_group_barrier(SG_VMEM_R, 8, 0);
            __builtin_amdgcn_sched_group_barrier(SG_DS_R,   2, 0);
            __builtin_amdgcn_sched_group_barrier(SG_MFMA,   4, 0);
        }
        __builtin_amdgcn_sched_group_barrier(SG_MFMA, 4, 0);
#endif

        #pragma unroll
        for (int r = 0; r < 8; ++r) {
            const int row = rhalf + r;
            const float xv = xlds[row * XSTR + t];
            float pg = ag[r] + xv * wgx + bgn;
            float pi = ai[r] + xv * wix + bin;
            float pf = af[r] + xv * wfx + bon;  // reference bug: f uses b_o
            float po = ao[r] + xv * wox + bfn;  // reference bug: o uses b_f
            float g  = ftanh(pg);
            float ig = fsigmoid(pi);
            float fg = fsigmoid(pf);
            float og = fsigmoid(po);
            float cnew = g * ig + creg[r] * fg;
            creg[r] = cnew;
            hn[row * HSTR + ncol] = (_Float16)(ftanh(cnew) * og);
        }

        __syncthreads();
        _Float16* tmp = hc; hc = hn; hn = tmp;
    }

    // final projection: out = h_T @ W_ph + b_p  (16 rows x 10 classes per block)
    if (tid < BM * 10) {
        const int row = tid / 10;
        const int cls = tid % 10;
        float s = bp[cls];
        for (int k = 0; k < HDIM; ++k)
            s += (float)hc[row * HSTR + k] * Wph[k * 10 + cls];
        out[(size_t)(rowbase + row) * 10 + cls] = s;
    }
}

extern "C" void kernel_launch(void* const* d_in, const int* in_sizes, int n_in,
                              void* d_out, int out_size, void* d_ws, size_t ws_size,
                              hipStream_t stream) {
    const float* x   = (const float*)d_in[0];
    const float* Wgx = (const float*)d_in[1];
    const float* Wgh = (const float*)d_in[2];
    const float* bg  = (const float*)d_in[3];
    const float* Wix = (const float*)d_in[4];
    const float* Wih = (const float*)d_in[5];
    const float* bi  = (const float*)d_in[6];
    const float* Wfx = (const float*)d_in[7];
    const float* Wfh = (const float*)d_in[8];
    const float* bf  = (const float*)d_in[9];
    const float* Wox = (const float*)d_in[10];
    const float* Woh = (const float*)d_in[11];
    const float* bo  = (const float*)d_in[12];
    const float* Wph = (const float*)d_in[13];
    const float* bp  = (const float*)d_in[14];
    _Float16* wswz = (_Float16*)d_ws;
    float* out = (float*)d_out;

    // 1) pre-swizzle recurrent weights into f16 WMMA B-fragments (512 KB in ws)
    lstm_prep_weights<<<dim3(4 * HDIM * HDIM / 256), dim3(256), 0, stream>>>(
        Wgh, Wih, Wfh, Woh, wswz);

    // 2) persistent scan: 64 blocks x 16 batch rows, 16 waves each
    const size_t shmem = (size_t)2 * BM * HSTR * 2   // h double buffer (f16)
                       + (size_t)BM * TSTEPS * 4;    // staged x slice (f32)
    lstm_persistent<<<dim3(1024 / BM), dim3(512), shmem, stream>>>(
        x, Wgx, bg, Wix, bi, Wfx, bf, Wox, bo, Wph, bp, wswz, out);
}
